// LocalMHA_82557861364357
// MI455X (gfx1250) — matrix operand order repaired and checked by reading, not compiled, not run on hardware
//
#include <hip/hip_runtime.h>
#include <stdint.h>

// Problem constants (fixed by the reference)
#define DMODEL 512
#define NHEAD  8
#define HEADD  64
#define WIN    64      // tokens per window (n = T / windows)
#define NWIN   256
#define BATCH  4
#define TLEN   16384

typedef __attribute__((ext_vector_type(16))) __bf16 v16bf;
typedef __attribute__((ext_vector_type(8)))  float  v8f;

union FragAB { uint4 u[2]; v16bf v; };

__device__ __forceinline__ uint16_t f2bf(float f) {
  uint32_t u = __builtin_bit_cast(uint32_t, f);
  u = (u + 0x7FFFu + ((u >> 16) & 1u)) >> 16;   // round-to-nearest-even
  return (uint16_t)u;
}
__device__ __forceinline__ float bf2f(uint16_t h) {
  uint32_t u = ((uint32_t)h) << 16;
  return __builtin_bit_cast(float, u);
}
__device__ __forceinline__ uint32_t pack2(float lo, float hi) {
  return (uint32_t)f2bf(lo) | ((uint32_t)f2bf(hi) << 16);
}

// ---- WMMA fragment loaders -------------------------------------------------
// A-matrix 16x32 bf16: lane<16 -> M=m0+lane, K = k0+[0..15];
// lane>=16 -> K = k0+16+[0..15]. Two ds_load_b128.
__device__ __forceinline__ FragAB load_a(const uint16_t* base, int stride,
                                         int m0, int k0, int lane) {
  int row = m0 + (lane & 15);
  int kb  = k0 + ((lane >> 4) << 4);
  FragAB f;
  f.u[0] = *(const uint4*)(base + row * stride + kb);
  f.u[1] = *(const uint4*)(base + row * stride + kb + 8);
  return f;
}
// B-matrix 32x16 bf16 where the SOURCE buffer is laid out [N][K] row-major
// (i.e. already "transposed" relative to B): lane<16 -> N=n0+lane, K=k0+[0..15];
// lane>=16 -> K=k0+16+[0..15]. Two ds_load_b128 (contiguous along K).
__device__ __forceinline__ FragAB load_bt(const uint16_t* base, int stride,
                                          int n0, int k0, int lane) {
  int row = n0 + (lane & 15);
  int kb  = k0 + ((lane >> 4) << 4);
  FragAB f;
  f.u[0] = *(const uint4*)(base + row * stride + kb);
  f.u[1] = *(const uint4*)(base + row * stride + kb + 8);
  return f;
}
// Pre-swizzled weight B-fragment: 256 u32 per tile, offset = lane*8 + v.
__device__ __forceinline__ FragAB load_bw(const uint32_t* fragtile, int lane) {
  FragAB f;
  f.u[0] = *(const uint4*)(fragtile + lane * 8);
  f.u[1] = *(const uint4*)(fragtile + lane * 8 + 4);
  return f;
}

__device__ __forceinline__ v8f wmma_bf16(const FragAB& a, const FragAB& b, v8f c) {
  return __builtin_amdgcn_wmma_f32_16x16x32_bf16(false, a.v, false, b.v,
                                                 (short)0, c, false, false);
}

// ---- Prep: fp32 weights -> bf16 WMMA B-fragments in workspace --------------
// Layout: frags[mat*131072 + (ktile*32 + ctile)*256 + lane*8 + v]
// Contiguous k order (same as load_a): lane<16 holds K=2v,2v+1 ; lane>=16 holds K=16+2v,17+2v.
__global__ void prep_weight_frags(const float* __restrict__ wq,
                                  const float* __restrict__ wk,
                                  const float* __restrict__ wv,
                                  const float* __restrict__ wo,
                                  uint32_t* __restrict__ frags) {
  int idx  = blockIdx.x * blockDim.x + threadIdx.x;   // 4*16*32*256 = 524288
  int v    = idx & 7;
  int lane = (idx >> 3) & 31;
  int ct   = (idx >> 8) & 31;
  int kt   = (idx >> 13) & 15;
  int mat  = idx >> 17;
  const float* w = (mat == 0) ? wq : (mat == 1) ? wk : (mat == 2) ? wv : wo;
  int n = ct * 16 + (lane & 15);
  int k = kt * 32 + ((lane >> 4) << 4) + 2 * v;
  frags[idx] = pack2(w[k * DMODEL + n], w[(k + 1) * DMODEL + n]);
}

// ---- Fused LN + QKV + RoPE + windowed attention + out-proj + residual ------
__global__ void __launch_bounds__(256, 1)
local_mha_kernel(const float* __restrict__ x, const float* __restrict__ ln_scale,
                 const float* __restrict__ ln_bias,
                 const uint32_t* __restrict__ wfrag, float* __restrict__ out) {
  extern __shared__ __align__(16) char smem[];
  uint16_t* XN = (uint16_t*)smem;       // 64x512 bf16 x-hat; reused as O later
  uint16_t* Qb = XN + 32768;            // 64x512 bf16, row-major [token][dim]
  uint16_t* Kb = Qb + 32768;            // 64x512 bf16, row-major [token][dim]
  uint16_t* VT = Kb + 32768;            // 8 heads x [64 e][64 token] (transposed V)
  uint16_t* Pb = VT + 32768;            // 8 heads x [16][64] probs scratch

  const int tid  = threadIdx.x;
  const int lane = tid & 31;
  const int wave = tid >> 5;
  const int bid  = blockIdx.x;
  const int b    = bid >> 8;
  const int w    = bid & 255;
  const long rowbase = (long)b * TLEN + (long)w * WIN;

  // ---------------- LayerNorm (wave handles 8 tokens) ----------------
  for (int i = 0; i < 8; ++i) {
    int m = wave * 8 + i;
    const float4* xr = (const float4*)(x + (rowbase + m) * DMODEL);
    float4 vals[4];
    float s = 0.f, ss = 0.f;
    for (int j = 0; j < 4; ++j) {
      float4 vv = xr[lane + j * 32];
      vals[j] = vv;
      s  += vv.x + vv.y + vv.z + vv.w;
      ss += vv.x * vv.x + vv.y * vv.y + vv.z * vv.z + vv.w * vv.w;
    }
    for (int d = 16; d >= 1; d >>= 1) {
      s  += __shfl_xor(s, d, 32);
      ss += __shfl_xor(ss, d, 32);
    }
    float mean = s * (1.f / 512.f);
    float var  = ss * (1.f / 512.f) - mean * mean;
    float rstd = rsqrtf(var + 1e-6f);
    for (int j = 0; j < 4; ++j) {
      int c4 = lane + j * 32;
      float4 g  = ((const float4*)ln_scale)[c4];
      float4 bb = ((const float4*)ln_bias)[c4];
      float4 vv = vals[j];
      uint2 p;
      p.x = pack2((vv.x - mean) * rstd * g.x + bb.x,
                  (vv.y - mean) * rstd * g.y + bb.y);
      p.y = pack2((vv.z - mean) * rstd * g.z + bb.z,
                  (vv.w - mean) * rstd * g.w + bb.w);
      *(uint2*)(XN + m * DMODEL + c4 * 4) = p;
    }
  }
  __syncthreads();

  // ---------------- QKV projections (3 x 4 x 32 output tiles) ----------------
  for (int t = wave; t < 384; t += 8) {
    int mat = t >> 7;
    int r   = t & 127;
    int mt  = r >> 5, ct = r & 31;
    int m0 = mt * 16, c0 = ct * 16;
    v8f acc = {};
    const uint32_t* fb = wfrag + mat * 131072 + ct * 256;
    for (int kc = 0; kc < 16; ++kc) {
      FragAB a  = load_a(XN, DMODEL, m0, kc * 32, lane);
      FragAB bf = load_bw(fb + kc * 8192, lane);
      acc = wmma_bf16(a, bf, acc);
    }
    int half = lane >> 4, col = lane & 15;
    if (mat < 2) {
      uint16_t* dst = mat ? Kb : Qb;
      for (int rr = 0; rr < 8; ++rr)
        dst[(m0 + rr + 8 * half) * DMODEL + c0 + col] = f2bf(acc[rr]);
    } else {
      int h = ct >> 2;
      int e = (ct & 3) * 16 + col;
      uint4 p;
      p.x = pack2(acc[0], acc[1]);
      p.y = pack2(acc[2], acc[3]);
      p.z = pack2(acc[4], acc[5]);
      p.w = pack2(acc[6], acc[7]);
      *(uint4*)(VT + (h * 64 + e) * 64 + m0 + 8 * half) = p;   // transposed store
    }
  }
  __syncthreads();

  // ---------------- Rotary on Q and K (in-place, pair-owned) ----------------
  const float NEG_LOG1E4_OVER_32 = -9.210340371976184f / 32.f;
  for (int i = tid; i < 2 * WIN * NHEAD * 32; i += 256) {
    int matsel = i >> 14;
    int rem    = i & 16383;
    int token  = rem >> 8;
    int h      = (rem >> 5) & 7;
    int j      = rem & 31;
    uint16_t* buf = matsel ? Kb : Qb;
    int i1 = token * DMODEL + h * 64 + j;
    float q1 = bf2f(buf[i1]), q2 = bf2f(buf[i1 + 32]);
    float ang = (float)token * __expf((float)j * NEG_LOG1E4_OVER_32);
    float sn, cs;
    __sincosf(ang, &sn, &cs);
    buf[i1]      = f2bf(q1 * cs - q2 * sn);
    buf[i1 + 32] = f2bf(q2 * cs + q1 * sn);
  }
  __syncthreads();

  // ---------------- Attention: wave == head ----------------
  {
    int h = wave;
    uint16_t* Ph = Pb + h * (16 * 64);
    const uint16_t* Vh = VT + h * (64 * 64);
    int half = lane >> 4, col = lane & 15;
    for (int mt = 0; mt < 4; ++mt) {
      int m0 = mt * 16;
      FragAB aq0 = load_a(Qb, DMODEL, m0, h * 64, lane);
      FragAB aq1 = load_a(Qb, DMODEL, m0, h * 64 + 32, lane);
      v8f sarr[4] = {{}, {}, {}, {}};
      for (int nt = 0; nt < 4; ++nt) {
        FragAB b0 = load_bt(Kb, DMODEL, nt * 16, h * 64, lane);
        FragAB b1 = load_bt(Kb, DMODEL, nt * 16, h * 64 + 32, lane);
        sarr[nt] = wmma_bf16(aq0, b0, sarr[nt]);
        sarr[nt] = wmma_bf16(aq1, b1, sarr[nt]);
      }
      // softmax over 64 key positions (4 tiles horizontally; width-16 shfl keeps
      // the two lane-halves' distinct rows independent)
      float mx[8], sm[8];
      for (int rr = 0; rr < 8; ++rr) {
        float m = fmaxf(fmaxf(sarr[0][rr], sarr[1][rr]),
                        fmaxf(sarr[2][rr], sarr[3][rr]));
        for (int d = 8; d >= 1; d >>= 1) m = fmaxf(m, __shfl_xor(m, d, 16));
        mx[rr] = m;
        sm[rr] = 0.f;
      }
      for (int nt = 0; nt < 4; ++nt)
        for (int rr = 0; rr < 8; ++rr) {
          float e = __expf((sarr[nt][rr] - mx[rr]) * 0.125f);  // 1/sqrt(64)
          sarr[nt][rr] = e;
          sm[rr] += e;
        }
      for (int rr = 0; rr < 8; ++rr) {
        float s = sm[rr];
        for (int d = 8; d >= 1; d >>= 1) s += __shfl_xor(s, d, 16);
        sm[rr] = 1.f / s;
      }
      // C-layout -> A-layout via per-head LDS scratch
      for (int nt = 0; nt < 4; ++nt)
        for (int rr = 0; rr < 8; ++rr)
          Ph[(rr + 8 * half) * 64 + nt * 16 + col] = f2bf(sarr[nt][rr] * sm[rr]);
      // O(16x64) = P(16x64) @ V_h(64x64), B-fragments from transposed V
      FragAB ap0 = load_a(Ph, 64, 0, 0, lane);
      FragAB ap1 = load_a(Ph, 64, 0, 32, lane);
      for (int et = 0; et < 4; ++et) {
        v8f o = {};
        FragAB b0 = load_bt(Vh, 64, et * 16, 0, lane);
        FragAB b1 = load_bt(Vh, 64, et * 16, 32, lane);
        o = wmma_bf16(ap0, b0, o);
        o = wmma_bf16(ap1, b1, o);
        for (int rr = 0; rr < 8; ++rr)
          XN[(m0 + rr + 8 * half) * DMODEL + h * 64 + et * 16 + col] = f2bf(o[rr]);
      }
    }
  }
  __syncthreads();

  // ---------------- Output projection + residual ----------------
  for (int t = wave; t < 128; t += 8) {
    int mt = t >> 5, ct = t & 31;
    int m0 = mt * 16, c0 = ct * 16;
    v8f acc = {};
    const uint32_t* fb = wfrag + 3 * 131072 + ct * 256;
    for (int kc = 0; kc < 16; ++kc) {
      FragAB a  = load_a(XN, DMODEL, m0, kc * 32, lane);
      FragAB bf = load_bw(fb + kc * 8192, lane);
      acc = wmma_bf16(a, bf, acc);
    }
    int half = lane >> 4, col = lane & 15;
    for (int rr = 0; rr < 8; ++rr) {
      long gi = (rowbase + m0 + rr + 8 * half) * (long)DMODEL + c0 + col;
      out[gi] = x[gi] + acc[rr];
    }
  }
}

extern "C" void kernel_launch(void* const* d_in, const int* in_sizes, int n_in,
                              void* d_out, int out_size, void* d_ws, size_t ws_size,
                              hipStream_t stream) {
  const float* x   = (const float*)d_in[0];
  const float* lns = (const float*)d_in[1];
  const float* lnb = (const float*)d_in[2];
  const float* wq  = (const float*)d_in[3];
  const float* wk  = (const float*)d_in[4];
  const float* wv  = (const float*)d_in[5];
  const float* wo  = (const float*)d_in[6];
  uint32_t* frags  = (uint32_t*)d_ws;          // 4 * 512 KB = 2 MB
  float* out       = (float*)d_out;

  prep_weight_frags<<<dim3(2048), dim3(256), 0, stream>>>(wq, wk, wv, wo, frags);

  size_t smem = (size_t)(32768 * 4 + 8 * 16 * 64) * sizeof(uint16_t); // 272 KB
  local_mha_kernel<<<dim3(BATCH * NWIN), dim3(256), smem, stream>>>(
      x, lns, lnb, frags, out);
}
